// EfficientAtt_43774306681437
// MI455X (gfx1250) — matrix operand order repaired and checked by reading, not compiled, not run on hardware
//
#include <hip/hip_runtime.h>
#include <math.h>

typedef __attribute__((ext_vector_type(16))) _Float16 v16h;
typedef __attribute__((ext_vector_type(8)))  float    v8f;

union V16HU { v16h v; _Float16 e[16]; };

__device__ __forceinline__ float hswish(float x){
  return x * fminf(fmaxf(x + 3.0f, 0.0f), 6.0f) * (1.0f/6.0f);
}

// A-tile 16x32 (f16): lane<16 -> K base 0 (elems 0-15 => K0-15)
//                     lane>=16 -> K base 16 (K16-31). M = lane&15.
__device__ __forceinline__ v16h load_a16(const float* __restrict__ A, int ld, int lane){
  int m  = lane & 15;
  int kb = (lane < 16) ? 0 : 16;
  const float* p = A + m*ld + kb;
  V16HU u;
#pragma unroll
  for (int i=0;i<8;i++) u.e[i]   = (_Float16)p[i];
#pragma unroll
  for (int i=0;i<8;i++) u.e[8+i] = (_Float16)p[8+i];
  return u.v;
}

// B-tile 32x16 (f16), source row-major [n][k]: N = lane&15, K = base(0/16)+i
__device__ __forceinline__ v16h load_b16_rowT(const float* __restrict__ X, int ld, int lane){
  int n  = lane & 15;
  int kb = (lane < 16) ? 0 : 16;
  const float* p = X + n*ld + kb;
  V16HU u;
#pragma unroll
  for (int i=0;i<16;i++) u.e[i] = (_Float16)p[i];
  return u.v;
}

// B-tile 32x16 (f16), source channel-major [k][tok] with row stride 4096
__device__ __forceinline__ v16h load_b16_chm(const float* __restrict__ S, int lane){
  int n  = lane & 15;
  int kb = (lane < 16) ? 0 : 16;
  const float* p = S + kb*4096 + n;
  V16HU u;
#pragma unroll
  for (int i=0;i<16;i++) u.e[i] = (_Float16)p[i*4096];
  return u.v;
}

// CDNA5 async copy: global -> LDS without VGPR round-trip (ASYNCcnt tracked)
__device__ __forceinline__ void async_g2lds_b128(unsigned lds_byte_off, const float* gptr){
  asm volatile("global_load_async_to_lds_b128 %0, %1, off"
               :: "v"(lds_byte_off), "v"(gptr) : "memory");
}
__device__ __forceinline__ void wait_async0(){
  asm volatile("s_wait_asynccnt 0" ::: "memory");
}

// ---------------- 1) qkv GEMM: out[g][b][c][pix] = sum_k W[g*320+c][k]*X[b*4096+pix][k]
__global__ __launch_bounds__(256) void qkv_gemm(const float* __restrict__ X,
                                                const float* __restrict__ Wq,
                                                float* __restrict__ out){
  int tid = threadIdx.x, lane = tid & 31, w = tid >> 5;
  int row0 = blockIdx.x*64 + (w & 1)*32;     // cout (of 960)
  int tok0 = blockIdx.y*256 + (w >> 1)*64;   // token (of 32768)
  v8f acc[2][4];
#pragma unroll
  for (int i=0;i<2;i++)
#pragma unroll
    for (int j=0;j<4;j++){ v8f z = {0,0,0,0,0,0,0,0}; acc[i][j]=z; }
  for (int k0=0;k0<320;k0+=32){
    v16h a0 = load_a16(Wq + (row0     )*320 + k0, 320, lane);
    v16h a1 = load_a16(Wq + (row0 + 16)*320 + k0, 320, lane);
    v16h b[4];
#pragma unroll
    for (int j=0;j<4;j++) b[j] = load_b16_rowT(X + (tok0 + j*16)*320 + k0, 320, lane);
#pragma unroll
    for (int j=0;j<4;j++){
      acc[0][j] = __builtin_amdgcn_wmma_f32_16x16x32_f16(false, a0, false, b[j], (short)0, acc[0][j], false, false);
      acc[1][j] = __builtin_amdgcn_wmma_f32_16x16x32_f16(false, a1, false, b[j], (short)0, acc[1][j], false, false);
    }
  }
  int mB = (lane < 16) ? 0 : 8, nO = lane & 15;
#pragma unroll
  for (int i=0;i<2;i++)
#pragma unroll
    for (int j=0;j<4;j++)
#pragma unroll
      for (int r=0;r<8;r++){
        int cout = row0 + i*16 + mB + r;
        int tok  = tok0 + j*16 + nO;
        int g  = cout / 320;
        int c  = cout - g*320;
        int bi = tok >> 12;
        int pix = tok & 4095;
        out[((g*8 + bi)*320 + c)*4096 + pix] = acc[i][j][r];
      }
}

// ---------------- generic depthwise conv (64x64 images, chan-major), optional bias
__global__ __launch_bounds__(256) void dw_conv(const float* __restrict__ src, int sBS, int sChOff,
                                               float* __restrict__ dst, int dBS, int dChOff,
                                               int nChan, int ks, int pad,
                                               const float* __restrict__ w,
                                               const float* __restrict__ bias, int aggMode){
  int bc = blockIdx.x;
  int bi = bc / nChan;
  int c  = bc - bi*nChan;
  int pix = blockIdx.y*256 + threadIdx.x;
  int y0 = pix >> 6, x0 = pix & 63;
  const float* sb;
  if (aggMode){                  // agg0: channel j = g*64+cc maps to qkv[g][bi][256+cc]
    int g = c >> 6, cc = c & 63;
    sb = src + ((g*8 + bi)*320 + 256 + cc)*4096;
  } else {
    sb = src + bi*sBS + (sChOff + c)*4096;
  }
  // prefetch the conv window rows for this pixel block (global_prefetch_b8)
  __builtin_prefetch(sb + pix, 0, 1);
  const float* wc = w + c*ks*ks;
  float acc = bias ? bias[c] : 0.0f;
  for (int ky=0; ky<ks; ky++){
    int yy = y0 + ky - pad;
    if (yy < 0 || yy > 63) continue;
    for (int kx=0; kx<ks; kx++){
      int xx = x0 + kx - pad;
      if (xx < 0 || xx > 63) continue;
      acc += wc[ky*ks+kx] * sb[yy*64 + xx];
    }
  }
  dst[bi*dBS + (dChOff + c)*4096 + pix] = acc;
}

// ---------------- 3) x0: BN+hswish of segment 0 -> xc channels 0..63
__global__ __launch_bounds__(256) void x0_bn(const float* __restrict__ qkv,
                                             const float* __restrict__ g, const float* __restrict__ b,
                                             const float* __restrict__ m, const float* __restrict__ v,
                                             float* __restrict__ xc){
  int idx = blockIdx.x*256 + threadIdx.x;    // 24*64*4096
  int gb = idx >> 18;
  int r  = idx & 262143;
  int c  = r >> 12;
  int pix = r & 4095;
  float s = g[c]*rsqrtf(v[c] + 1e-5f);
  float val = qkv[(gb*320 + c)*4096 + pix]*s + (b[c] - m[c]*s);
  xc[gb*256*4096 + c*4096 + pix] = hswish(val);
}

// ---------------- 4) pointwise 64x64 WMMA + BN + hswish -> xc channels (z+1)*64 ..
__global__ __launch_bounds__(256) void pw_gemm(const float* __restrict__ dw1, const float* __restrict__ dw2, const float* __restrict__ dw3,
                                               const float* __restrict__ pw1, const float* __restrict__ pw2, const float* __restrict__ pw3,
                                               const float* __restrict__ g1, const float* __restrict__ b1, const float* __restrict__ m1, const float* __restrict__ v1,
                                               const float* __restrict__ g2, const float* __restrict__ b2, const float* __restrict__ m2, const float* __restrict__ v2,
                                               const float* __restrict__ g3, const float* __restrict__ b3, const float* __restrict__ m3, const float* __restrict__ v3,
                                               float* __restrict__ xc){
  int z = blockIdx.z;
  const float* dwb = (z==0)?dw1:((z==1)?dw2:dw3);
  const float* pwb = (z==0)?pw1:((z==1)?pw2:pw3);
  const float* bg  = (z==0)?g1:((z==1)?g2:g3);
  const float* bb  = (z==0)?b1:((z==1)?b2:b3);
  const float* bm  = (z==0)?m1:((z==1)?m2:m3);
  const float* bv  = (z==0)?v1:((z==1)?v2:v3);
  int gb = blockIdx.y;
  const float* src = dwb + gb*64*4096;
  int tid = threadIdx.x, lane = tid & 31, w = tid >> 5;
  int c0   = (w & 1)*32;
  int tok0 = blockIdx.x*128 + (w >> 1)*32;
  v8f acc[2][2];
#pragma unroll
  for (int i=0;i<2;i++)
#pragma unroll
    for (int j=0;j<2;j++){ v8f zz = {0,0,0,0,0,0,0,0}; acc[i][j]=zz; }
  for (int k0=0;k0<64;k0+=32){
    v16h a0 = load_a16(pwb + (c0     )*64 + k0, 64, lane);
    v16h a1 = load_a16(pwb + (c0 + 16)*64 + k0, 64, lane);
    v16h q0 = load_b16_chm(src + k0*4096 + tok0,      lane);
    v16h q1 = load_b16_chm(src + k0*4096 + tok0 + 16, lane);
    acc[0][0] = __builtin_amdgcn_wmma_f32_16x16x32_f16(false, a0, false, q0, (short)0, acc[0][0], false, false);
    acc[0][1] = __builtin_amdgcn_wmma_f32_16x16x32_f16(false, a0, false, q1, (short)0, acc[0][1], false, false);
    acc[1][0] = __builtin_amdgcn_wmma_f32_16x16x32_f16(false, a1, false, q0, (short)0, acc[1][0], false, false);
    acc[1][1] = __builtin_amdgcn_wmma_f32_16x16x32_f16(false, a1, false, q1, (short)0, acc[1][1], false, false);
  }
  int mB = (lane < 16) ? 0 : 8, nO = lane & 15;
#pragma unroll
  for (int i=0;i<2;i++)
#pragma unroll
    for (int j=0;j<2;j++)
#pragma unroll
      for (int r=0;r<8;r++){
        int c   = c0 + i*16 + mB + r;
        int tok = tok0 + j*16 + nO;
        float s = bg[c]*rsqrtf(bv[c] + 1e-5f);
        float val = acc[i][j][r]*s + (bb[c] - bm[c]*s);
        xc[gb*256*4096 + ((z+1)*64 + c)*4096 + tok] = hswish(val);
      }
}

// ---------------- 5) agg0 pointwise 192->64 + LN(64) + hswish -> concat ch 256..319
__global__ __launch_bounds__(256) void agg0_pw_ln(const float* __restrict__ dwa,
                                                  const float* __restrict__ pw,
                                                  const float* __restrict__ lng, const float* __restrict__ lnb,
                                                  float* __restrict__ cat){
  __shared__ float sv[4][64];
  int c  = threadIdx.x & 63;
  int lt = threadIdx.x >> 6;
  int pix = blockIdx.x*4 + lt;
  int b = blockIdx.y;
  const float* in = dwa + b*192*4096 + pix;
  float val = 0.0f;
  for (int cin=0; cin<192; cin++) val += pw[c*192 + cin] * in[cin*4096];
  sv[lt][c] = val;
  __syncthreads();
  float mu = 0.0f;
#pragma unroll
  for (int j=0;j<64;j++) mu += sv[lt][j];
  mu *= (1.0f/64.0f);
  float var = 0.0f;
#pragma unroll
  for (int j=0;j<64;j++){ float d = sv[lt][j] - mu; var += d*d; }
  var *= (1.0f/64.0f);
  float y = (val - mu)*rsqrtf(var + 1e-5f)*lng[c] + lnb[c];
  cat[(b*4096 + pix)*320 + 256 + c] = hswish(y);
}

// ---------------- 6) softmax over 4096 tokens per (b, channel) row of k, in place
__global__ __launch_bounds__(256) void softmax_rows(float* __restrict__ kbuf){
  __shared__ float red[256];
  float* p = kbuf + (size_t)blockIdx.x*4096;
  int tid = threadIdx.x;
  float vals[16];
  float lm = -1e30f;
#pragma unroll
  for (int i=0;i<16;i++){ vals[i] = p[tid + 256*i]; lm = fmaxf(lm, vals[i]); }
  red[tid] = lm; __syncthreads();
  for (int s=128; s>0; s>>=1){ if (tid < s) red[tid] = fmaxf(red[tid], red[tid+s]); __syncthreads(); }
  float gm = red[0]; __syncthreads();
  float ls = 0.0f;
#pragma unroll
  for (int i=0;i<16;i++){ vals[i] = expf(vals[i] - gm); ls += vals[i]; }
  red[tid] = ls; __syncthreads();
  for (int s=128; s>0; s>>=1){ if (tid < s) red[tid] += red[tid+s]; __syncthreads(); }
  float inv = 1.0f / red[0];
#pragma unroll
  for (int i=0;i<16;i++) p[tid + 256*i] = vals[i]*inv;
}

// ---------------- 7) kv[b][h][ck][cv] = sum_n ksm[b][h*32+ck][n] * v[b][h*32+cv][n]
// k/v tiles staged into LDS with CDNA5 async global->LDS DMA (ASYNCcnt).
__global__ __launch_bounds__(256) void kv_gemm(const float* __restrict__ ksm,
                                               const float* __restrict__ vv,
                                               float* __restrict__ kv){
  __shared__ float sk[32*128];
  __shared__ float sv[32*128];
  int b  = blockIdx.x >> 3;
  int hd = blockIdx.x & 7;
  const float* kb = ksm + (b*256 + hd*32)*4096;
  const float* vb = vv  + (b*256 + hd*32)*4096;
  int tid = threadIdx.x;
  int ck  = tid >> 3;
  int cvb = (tid*4) & 31;
  unsigned skBase = (unsigned)(size_t)(&sk[0]);
  unsigned svBase = (unsigned)(size_t)(&sv[0]);
  float acc[4] = {0,0,0,0};
  for (int t0=0; t0<4096; t0+=128){
#pragma unroll
    for (int r=0;r<4;r++){
      int li  = (r*256 + tid)*4;        // float index within 32x128 tile (b128 per lane)
      int chn = li >> 7, tk = li & 127;
      unsigned loff = (unsigned)li*4u;  // byte offset
      async_g2lds_b128(skBase + loff, kb + chn*4096 + t0 + tk);
      async_g2lds_b128(svBase + loff, vb + chn*4096 + t0 + tk);
    }
    wait_async0();
    __syncthreads();
    for (int t=0;t<128;t++){
      float kl = sk[ck*128 + t];
#pragma unroll
      for (int r=0;r<4;r++) acc[r] += kl * sv[(cvb+r)*128 + t];
    }
    __syncthreads();
  }
#pragma unroll
  for (int r=0;r<4;r++) kv[(b*8+hd)*1024 + ck*32 + cvb + r] = acc[r];
}

// ---------------- 9) eff = q·kv fused with crpe = q*conv_v -> concat ch 0..255 (token-major)
__global__ __launch_bounds__(256) void combine_attn(const float* __restrict__ q,
                                                    const float* __restrict__ kv,
                                                    const float* __restrict__ convv,
                                                    float* __restrict__ cat){
  __shared__ float skv[32][32];
  int b = blockIdx.z, hd = blockIdx.y;
  int tok = blockIdx.x*256 + threadIdx.x;
#pragma unroll
  for (int i=0;i<4;i++){
    int li = i*256 + threadIdx.x;
    skv[li>>5][li&31] = kv[(b*8+hd)*1024 + li];
  }
  __syncthreads();
  const float* qb = q     + (b*256 + hd*32)*4096 + tok;
  const float* cb = convv + (b*256 + hd*32)*4096 + tok;
  float qr[32];
#pragma unroll
  for (int ck=0;ck<32;ck++) qr[ck] = qb[ck*4096];
  float* ob = cat + (b*4096 + tok)*320 + hd*32;
  const float scale = 0.15811388300841898f;  // (320/8)^-0.5
#pragma unroll
  for (int cv=0;cv<32;cv++){
    float acc = 0.0f;
#pragma unroll
    for (int ck=0;ck<32;ck++) acc += qr[ck]*skv[ck][cv];
    ob[cv] = scale*acc + qr[cv]*cb[cv*4096];
  }
}

// ---------------- 10) proj GEMM: out[t][c] = sum_k cat[t][k]*Wp[c][k] + bias[c]
__global__ __launch_bounds__(256) void proj_gemm(const float* __restrict__ cat,
                                                 const float* __restrict__ Wp,
                                                 const float* __restrict__ bp,
                                                 float* __restrict__ out){
  int tid = threadIdx.x, lane = tid & 31, w = tid >> 5;
  int row0 = blockIdx.x*256 + w*32;   // tokens
  int c0   = blockIdx.y*64;           // couts
  v8f acc[2][4];
#pragma unroll
  for (int i=0;i<2;i++)
#pragma unroll
    for (int j=0;j<4;j++){ v8f z = {0,0,0,0,0,0,0,0}; acc[i][j]=z; }
  for (int k0=0;k0<320;k0+=32){
    v16h a0 = load_a16(cat + (row0     )*320 + k0, 320, lane);
    v16h a1 = load_a16(cat + (row0 + 16)*320 + k0, 320, lane);
    v16h b[4];
#pragma unroll
    for (int j=0;j<4;j++) b[j] = load_b16_rowT(Wp + (c0 + j*16)*320 + k0, 320, lane);
#pragma unroll
    for (int j=0;j<4;j++){
      acc[0][j] = __builtin_amdgcn_wmma_f32_16x16x32_f16(false, a0, false, b[j], (short)0, acc[0][j], false, false);
      acc[1][j] = __builtin_amdgcn_wmma_f32_16x16x32_f16(false, a1, false, b[j], (short)0, acc[1][j], false, false);
    }
  }
  int mB = (lane < 16) ? 0 : 8, nO = lane & 15;
#pragma unroll
  for (int i=0;i<2;i++)
#pragma unroll
    for (int j=0;j<4;j++)
#pragma unroll
      for (int r=0;r<8;r++){
        int tok = row0 + i*16 + mB + r;
        int c   = c0 + j*16 + nO;
        out[tok*320 + c] = acc[i][j][r] + bp[c];
      }
}

extern "C" void kernel_launch(void* const* d_in, const int* in_sizes, int n_in,
                              void* d_out, int out_size, void* d_ws, size_t ws_size,
                              hipStream_t stream){
  (void)in_sizes; (void)n_in; (void)out_size; (void)ws_size;
  const float* x       = (const float*)d_in[0];
  const float* qkv_w   = (const float*)d_in[1];
  const float* proj_w  = (const float*)d_in[2];
  const float* proj_b  = (const float*)d_in[3];
  const float* agg0_dw = (const float*)d_in[4];
  const float* agg0_pw = (const float*)d_in[5];
  const float* ln_g    = (const float*)d_in[6];
  const float* ln_b    = (const float*)d_in[7];
  const float* agg1_dw = (const float*)d_in[8];
  const float* agg1_pw = (const float*)d_in[9];
  const float* agg2_dw = (const float*)d_in[10];
  const float* agg2_pw = (const float*)d_in[11];
  const float* agg3_dw = (const float*)d_in[12];
  const float* agg3_pw = (const float*)d_in[13];
  const float* crpe_w3 = (const float*)d_in[14];
  const float* crpe_b3 = (const float*)d_in[15];
  const float* crpe_w5 = (const float*)d_in[16];
  const float* crpe_b5 = (const float*)d_in[17];
  const float* crpe_w7 = (const float*)d_in[18];
  const float* crpe_b7 = (const float*)d_in[19];
  const float* bn0_g = (const float*)d_in[22];
  const float* bn0_b = (const float*)d_in[23];
  const float* bn0_m = (const float*)d_in[24];
  const float* bn0_v = (const float*)d_in[25];
  const float* bn1_g = (const float*)d_in[26];
  const float* bn1_b = (const float*)d_in[27];
  const float* bn1_m = (const float*)d_in[28];
  const float* bn1_v = (const float*)d_in[29];
  const float* bn2_g = (const float*)d_in[30];
  const float* bn2_b = (const float*)d_in[31];
  const float* bn2_m = (const float*)d_in[32];
  const float* bn2_v = (const float*)d_in[33];
  const float* bn3_g = (const float*)d_in[34];
  const float* bn3_b = (const float*)d_in[35];
  const float* bn3_m = (const float*)d_in[36];
  const float* bn3_v = (const float*)d_in[37];
  float* out = (float*)d_out;

  float* ws = (float*)d_ws;
  float* qkv  = ws;                   // [3][8][320][4096]  = 31457280 floats
  float* dw1  = ws + 31457280;        // [24][64][4096]     =  6291456
  float* dw2  = ws + 37748736;        // [24][64][4096]
  float* dw3  = ws + 44040192;        // [24][64][4096]
  float* dwa  = ws + 50331648;        // [8][192][4096]     =  6291456
  float* xc   = ws + 56623104;        // [3][8][256][4096]  = 25165824 (q,k,v)
  float* kvb  = ws + 81788928;        // [8][8][32][32]     =    65536
  float* cvv  = ws;                   // conv_v reuses qkv region: [8][256][4096]
  float* cat  = ws + 8388608;         // concat reuses qkv region: [8][4096][320]
  float* kbuf = xc + 8388608;         // k region
  float* vbuf = xc + 16777216;        // v region

  // 1) qkv projection -> channel-major
  qkv_gemm<<<dim3(15,128), 256, 0, stream>>>(x, qkv_w, qkv);

  // 2) depthwise convs (branches 1..3 and agg0 merged)
  dw_conv<<<dim3(24*64,16), 256, 0, stream>>>(qkv, 320*4096,  64, dw1, 64*4096, 0,  64, 3, 1, agg1_dw, nullptr, 0);
  dw_conv<<<dim3(24*64,16), 256, 0, stream>>>(qkv, 320*4096, 128, dw2, 64*4096, 0,  64, 5, 2, agg2_dw, nullptr, 0);
  dw_conv<<<dim3(24*64,16), 256, 0, stream>>>(qkv, 320*4096, 192, dw3, 64*4096, 0,  64, 7, 3, agg3_dw, nullptr, 0);
  dw_conv<<<dim3(8*192,16), 256, 0, stream>>>(qkv, 0,          0, dwa, 192*4096, 0, 192, 3, 1, agg0_dw, nullptr, 1);

  // 3) x0 = hswish(BN(seg0))   (qkv region no longer needed after this)
  x0_bn<<<dim3(24576), 256, 0, stream>>>(qkv, bn0_g, bn0_b, bn0_m, bn0_v, xc);

  // 4) pointwise WMMA + BN + hswish for branches 1..3
  pw_gemm<<<dim3(32,24,3), 256, 0, stream>>>(dw1, dw2, dw3, agg1_pw, agg2_pw, agg3_pw,
                                             bn1_g, bn1_b, bn1_m, bn1_v,
                                             bn2_g, bn2_b, bn2_m, bn2_v,
                                             bn3_g, bn3_b, bn3_m, bn3_v, xc);

  // 5) agg0 pointwise + LN + hswish -> concat channels 256..319
  agg0_pw_ln<<<dim3(1024,8), 256, 0, stream>>>(dwa, agg0_pw, ln_g, ln_b, cat);

  // 6) softmax over tokens on k (in place)
  softmax_rows<<<dim3(2048), 256, 0, stream>>>(kbuf);

  // 7) kv = ksm^T v per (b, head) with async global->LDS staging
  kv_gemm<<<dim3(64), 256, 0, stream>>>(kbuf, vbuf, kvb);

  // 8) CRPE depthwise convs on v (with bias)
  dw_conv<<<dim3(8*32, 16), 256, 0, stream>>>(vbuf, 256*4096,  0, cvv, 256*4096,  0,  32, 3, 1, crpe_w3, crpe_b3, 0);
  dw_conv<<<dim3(8*64, 16), 256, 0, stream>>>(vbuf, 256*4096, 32, cvv, 256*4096, 32,  64, 5, 2, crpe_w5, crpe_b5, 0);
  dw_conv<<<dim3(8*160,16), 256, 0, stream>>>(vbuf, 256*4096, 96, cvv, 256*4096, 96, 160, 7, 3, crpe_w7, crpe_b7, 0);

  // 9) scale*eff + q*conv_v -> concat channels 0..255 (token-major)
  combine_attn<<<dim3(16,8,8), 256, 0, stream>>>(xc, kvb, cvv, cat);

  // 10) output projection + bias
  proj_gemm<<<dim3(128,5), 256, 0, stream>>>(cat, proj_w, proj_b, out);
}
